// EGConvDescriptors_16956530885123
// MI455X (gfx1250) — matrix operand order repaired and checked by reading, not compiled, not run on hardware
//
#include <hip/hip_runtime.h>
#include <cstdint>
#include <cstddef>

#define N_NODES 100000
#define N_EDGES 200000
#define N_GRAPH 4096
#define HID     512
#define NLAYER  4
#define HEADS   8
#define BASESN  8
#define FH      64
#define DESCN   200
#define COMBN   192   // HEADS*BASESN*3
#define K1      27
#define EPSI    1e-5f

typedef __bf16 bf16;
typedef __attribute__((ext_vector_type(16))) __bf16 v16bf;
typedef __attribute__((ext_vector_type(8)))  float  v8f;

struct U128 { unsigned int x, y, z, w; };
struct U256 { U128 lo, hi; };

__device__ __forceinline__ v16bf make_frag(U128 lo, U128 hi) {
    U256 p{lo, hi};
    return __builtin_bit_cast(v16bf, p);
}

// ---- CDNA5 async global->LDS path (guarded; falls back to reg staging) ----
#if __has_builtin(__builtin_amdgcn_global_load_async_to_lds_b128) && \
    __has_builtin(__builtin_amdgcn_s_wait_asynccnt)
#define HAS_ASYNC_LDS 1
typedef __attribute__((__vector_size__(4 * sizeof(int)))) int v4i_ls;
typedef __attribute__((address_space(1))) v4i_ls* g_v4i_p;
typedef __attribute__((address_space(3))) v4i_ls* l_v4i_p;
__device__ __forceinline__ void async_copy16(const bf16* g, bf16* l) {
    __builtin_amdgcn_global_load_async_to_lds_b128(
        (g_v4i_p)(v4i_ls*)(void*)g,
        (l_v4i_p)(v4i_ls*)(void*)l,
        0, 0);
}
#else
#define HAS_ASYNC_LDS 0
#endif

__device__ __forceinline__ void stage_wait_barrier() {
#if HAS_ASYNC_LDS
    __builtin_amdgcn_s_wait_asynccnt(0);
#endif
    __syncthreads();
}

// ---------------- weight conversion ----------------
__global__ void k_cvt_bf16(const float* __restrict__ s, bf16* __restrict__ d, long n) {
    long i = (long)blockIdx.x * blockDim.x + threadIdx.x;
    if (i < n) d[i] = (bf16)s[i];
}

// ---------------- lin1: [N,27] @ [27,512] + b ----------------
__global__ __launch_bounds__(256) void k_lin1(const float* __restrict__ x,
                                              const float* __restrict__ W,
                                              const float* __restrict__ b,
                                              float* __restrict__ y) {
    __shared__ float xs[K1];
    int n = blockIdx.x;
    if (threadIdx.x < K1) xs[threadIdx.x] = x[(size_t)n * K1 + threadIdx.x];
    __syncthreads();
    for (int c = threadIdx.x; c < HID; c += 256) {
        float acc = b[c];
#pragma unroll
        for (int k = 0; k < K1; k++) acc += xs[k] * W[k * HID + c];
        y[(size_t)n * HID + c] = acc;
    }
}

// ---------------- BN statistics (deterministic 2-stage) ----------------
__global__ __launch_bounds__(256) void k_bn_stats(const float* __restrict__ x,
                                                  float* __restrict__ partial,
                                                  int rows, int cols) {
    int t = threadIdx.x;
    float s0 = 0.f, q0 = 0.f, s1 = 0.f, q1 = 0.f;
    for (int r = blockIdx.x; r < rows; r += gridDim.x) {
        const float* row = x + (size_t)r * cols;
        if (t < cols)        { float v = row[t];        s0 += v; q0 += v * v; }
        if (t + 256 < cols)  { float v = row[t + 256];  s1 += v; q1 += v * v; }
    }
    float* p = partial + (size_t)blockIdx.x * 2 * cols;
    if (t < cols)       { p[t] = s0;        p[cols + t] = q0; }
    if (t + 256 < cols) { p[t + 256] = s1;  p[cols + t + 256] = q1; }
}

__global__ void k_bn_fin(const float* __restrict__ partial, float* __restrict__ stats,
                         int nblk, int cols, float invR) {
    int c = blockIdx.x * blockDim.x + threadIdx.x;
    if (c >= cols) return;
    float s = 0.f, q = 0.f;
    for (int b = 0; b < nblk; b++) {
        s += partial[(size_t)b * 2 * cols + c];
        q += partial[(size_t)b * 2 * cols + cols + c];
    }
    float m = s * invR;
    float v = q * invR - m * m;
    stats[c] = m;
    stats[cols + c] = rsqrtf(v + EPSI);
}

// y = (resid? resid[i]:0) + relu(bn(x));  optional bf16 copy
__global__ void k_bn_apply(const float* __restrict__ x, const float* __restrict__ stats,
                           const float* __restrict__ gam, const float* __restrict__ bet,
                           const float* __restrict__ resid, float* __restrict__ y,
                           bf16* __restrict__ ybf, long total, int cols) {
    long i = (long)blockIdx.x * blockDim.x + threadIdx.x;
    if (i >= total) return;
    int c = (int)(i % cols);
    float v = (x[i] - stats[c]) * stats[cols + c] * gam[c] + bet[c];
    v = fmaxf(v, 0.f);
    if (resid) v += resid[i];
    y[i] = v;
    if (ybf) ybf[i] = (bf16)v;
}

// ---------------- WMMA bf16 GEMM ----------------
// C[M,Ncols] = A[M,K] @ B[K,Ncols] (+bias); block tile 256x64, double-buffered LDS.
// 8 waves, each wave: 32 rows x 64 cols = 8 WMMA accumulators; K step 32.
// A tile staged via GLOBAL_LOAD_ASYNC_TO_LDS_B128 when available (ASYNCcnt path).
__device__ __forceinline__ void stage_tiles(const bf16* __restrict__ A, const bf16* __restrict__ B,
                                            int M, int Ncols, int K, int rowBase, int ctile,
                                            int k0, bf16* __restrict__ As, bf16* __restrict__ Bs,
                                            int tid) {
    // A: 256 rows x 32 k; thread t stages its whole row (64B)
    int gr = rowBase + tid;
    bf16* la = As + tid * 32;
    if (gr < M) {
        const bf16* ga = A + (size_t)gr * K + k0;
#if HAS_ASYNC_LDS
        async_copy16(ga,      la);
        async_copy16(ga + 8,  la + 8);
        async_copy16(ga + 16, la + 16);
        async_copy16(ga + 24, la + 24);
#else
        const U128* pa = reinterpret_cast<const U128*>(ga);
        U128* pl = reinterpret_cast<U128*>(la);
        pl[0] = pa[0]; pl[1] = pa[1]; pl[2] = pa[2]; pl[3] = pa[3];
#endif
    } else {
        U128 z{0, 0, 0, 0};
        U128* pl = reinterpret_cast<U128*>(la);
        pl[0] = z; pl[1] = z; pl[2] = z; pl[3] = z;
    }
    // B: 32 k x 64 cols, stored K-transposed: Bs[col*32 + k]
    int bk = tid >> 3, bc = (tid & 7) * 8;
    const U128* pb = reinterpret_cast<const U128*>(B + (size_t)(k0 + bk) * Ncols + ctile + bc);
    union { U128 u; bf16 h[8]; } tb;
    tb.u = pb[0];
#pragma unroll
    for (int j = 0; j < 8; j++) Bs[(bc + j) * 32 + bk] = tb.h[j];
}

__global__ __launch_bounds__(256) void k_wmma_gemm_bf16(const bf16* __restrict__ A,
                                                        const bf16* __restrict__ B,
                                                        const float* __restrict__ bias,
                                                        float* __restrict__ C,
                                                        int M, int Ncols, int K) {
    __shared__ alignas(16) bf16 As[2][256 * 32];
    __shared__ alignas(16) bf16 Bs[2][64 * 32];
    const int tid  = threadIdx.x;
    const int wid  = tid >> 5;
    const int lane = tid & 31;
    const int n16  = lane & 15;
    const int kh   = lane >> 4;
    const int rowBase = blockIdx.x * 256;
    const int ctile   = blockIdx.y * 64;

    const v8f vz = {0.f, 0.f, 0.f, 0.f, 0.f, 0.f, 0.f, 0.f};
    v8f acc[2][4];
#pragma unroll
    for (int mt = 0; mt < 2; mt++)
#pragma unroll
        for (int t = 0; t < 4; t++) acc[mt][t] = vz;

    const int nk = K / 32;
    stage_tiles(A, B, M, Ncols, K, rowBase, ctile, 0, As[0], Bs[0], tid);
    stage_wait_barrier();

    for (int s = 0; s < nk; s++) {
        if (s + 1 < nk)
            stage_tiles(A, B, M, Ncols, K, rowBase, ctile, (s + 1) * 32,
                        As[(s + 1) & 1], Bs[(s + 1) & 1], tid);

        const bf16* Ac = As[s & 1];
        const bf16* Bc = Bs[s & 1];
        // B fragments: elems 0..7 = K kh*8..+7, elems 8..15 = K 16+kh*8..+7 (K-transposed LDS)
        v16bf bfg[4];
#pragma unroll
        for (int t = 0; t < 4; t++) {
            const int cl = t * 16 + n16;
            bfg[t] = make_frag(*reinterpret_cast<const U128*>(&Bc[cl * 32 + kh * 8]),
                               *reinterpret_cast<const U128*>(&Bc[cl * 32 + 16 + kh * 8]));
        }
#pragma unroll
        for (int mt = 0; mt < 2; mt++) {
            // A fragment: elems 0..7 = K kh*8..+7, elems 8..15 = K 16+kh*8..+7
            const int arl = wid * 32 + mt * 16 + n16;
            v16bf af = make_frag(*reinterpret_cast<const U128*>(&Ac[arl * 32 + kh * 8]),
                                 *reinterpret_cast<const U128*>(&Ac[arl * 32 + 16 + kh * 8]));
#pragma unroll
            for (int t = 0; t < 4; t++)
                acc[mt][t] = __builtin_amdgcn_wmma_f32_16x16x32_bf16(
                    false, af, false, bfg[t], (short)0, acc[mt][t], false, false);
        }
        stage_wait_barrier();
    }

    // C layout: lane half kh -> M = kh*8 + r, N = lane&15
#pragma unroll
    for (int mt = 0; mt < 2; mt++) {
#pragma unroll
        for (int t = 0; t < 4; t++) {
#pragma unroll
            for (int r = 0; r < 8; r++) {
                int grow = rowBase + wid * 32 + mt * 16 + kh * 8 + r;
                if (grow < M) {
                    int col = ctile + t * 16 + n16;
                    float v = acc[mt][t][r];
                    if (bias) v += bias[col];
                    C[(size_t)grow * Ncols + col] = v;
                }
            }
        }
    }
}

// ---------------- message passing ----------------
__device__ __forceinline__ void atomicMaxF(float* addr, float val) {
    if (val >= 0.f) atomicMax(reinterpret_cast<int*>(addr), __float_as_int(val));
    else            atomicMin(reinterpret_cast<unsigned int*>(addr), __float_as_uint(val));
}

__global__ void k_init_scatter(float* ssum, float* smax, float* cnt) {
    long i = (long)blockIdx.x * blockDim.x + threadIdx.x;
    if (i < (long)N_NODES * HID) { ssum[i] = 0.f; smax[i] = -3.0e38f; }
    if (i < N_NODES) cnt[i] = 0.f;
}

__global__ void k_scatter(const float* __restrict__ bases, const int* __restrict__ ei,
                          float* ssum, float* smax, float* cnt) {
    long idx = (long)blockIdx.x * blockDim.x + threadIdx.x;
    const long tot = (long)(N_EDGES + N_NODES) * (HID / 4);
    if (idx >= tot) return;
    int e  = (int)(idx >> 7);
    int c4 = (int)(idx & 127);
    int src, dst;
    if (e < N_EDGES) { src = ei[e]; dst = ei[N_EDGES + e]; }
    else             { src = dst = e - N_EDGES; }          // self loops
    float4 v = *reinterpret_cast<const float4*>(bases + (size_t)src * HID + c4 * 4);
    float* sp = ssum + (size_t)dst * HID + c4 * 4;
    float* mp = smax + (size_t)dst * HID + c4 * 4;
    atomicAdd(sp + 0, v.x); atomicAdd(sp + 1, v.y);
    atomicAdd(sp + 2, v.z); atomicAdd(sp + 3, v.w);
    atomicMaxF(mp + 0, v.x); atomicMaxF(mp + 1, v.y);
    atomicMaxF(mp + 2, v.z); atomicMaxF(mp + 3, v.w);
    if (c4 == 0) atomicAdd(cnt + dst, 1.f);
}

// per-node einsum: out[h*64+f] = sum_k w[h,k]*agg[k,f], k = a*8+b, agg(a,b,f) in {sum,mean,max}
__global__ __launch_bounds__(256) void k_combine(const float* __restrict__ ssum,
                                                 const float* __restrict__ smax,
                                                 const float* __restrict__ cnt,
                                                 const float* __restrict__ comb,
                                                 const float* __restrict__ cbias,
                                                 float* __restrict__ mout) {
    __shared__ float wsh[4][COMBN];
    int ln = threadIdx.x >> 6;
    int f  = threadIdx.x & 63;
    int node = blockIdx.x * 4 + ln;
    if (node < N_NODES) {
        for (int j = f; j < COMBN; j += 64) wsh[ln][j] = comb[(size_t)node * COMBN + j];
    }
    __syncthreads();
    if (node >= N_NODES) return;
    float rc = 1.f / fmaxf(cnt[node], 1.f);
    float acc[HEADS];
#pragma unroll
    for (int h = 0; h < HEADS; h++) acc[h] = cbias[h * FH + f];
    size_t base = (size_t)node * HID;
#pragma unroll
    for (int b = 0; b < BASESN; b++) {
        float sv = ssum[base + b * FH + f];
        float mv = smax[base + b * FH + f];
        float mn = sv * rc;
#pragma unroll
        for (int h = 0; h < HEADS; h++) {
            const float* w = &wsh[ln][h * 24];
            acc[h] += w[b] * sv + w[8 + b] * mn + w[16 + b] * mv;
        }
    }
#pragma unroll
    for (int h = 0; h < HEADS; h++) mout[base + h * FH + f] = acc[h];
}

// ---------------- graph pooling ----------------
__global__ void k_pool_init(float* gsum, float* gcnt) {
    long i = (long)blockIdx.x * blockDim.x + threadIdx.x;
    if (i < (long)N_GRAPH * HID) gsum[i] = 0.f;
    if (i < N_GRAPH) gcnt[i] = 0.f;
}

__global__ void k_pool(const float* __restrict__ h, const int* __restrict__ batch,
                       float* gsum, float* gcnt) {
    long idx = (long)blockIdx.x * blockDim.x + threadIdx.x;
    if (idx >= (long)N_NODES * (HID / 4)) return;
    int node = (int)(idx >> 7);
    int c4   = (int)(idx & 127);
    int g = batch[node];
    float4 v = *reinterpret_cast<const float4*>(h + (size_t)node * HID + c4 * 4);
    float* p = gsum + (size_t)g * HID + c4 * 4;
    atomicAdd(p + 0, v.x); atomicAdd(p + 1, v.y);
    atomicAdd(p + 2, v.z); atomicAdd(p + 3, v.w);
    if (c4 == 0) atomicAdd(gcnt + g, 1.f);
}

__global__ void k_pool_div(float* gsum, const float* __restrict__ gcnt) {
    long i = (long)blockIdx.x * blockDim.x + threadIdx.x;
    if (i >= (long)N_GRAPH * HID) return;
    gsum[i] /= fmaxf(gcnt[i / HID], 1.f);
}

// ---------------- small fp32 GEMM for the [4096, .] MLP tail ----------------
__global__ void k_gemm_f32(const float* __restrict__ A, const float* __restrict__ W,
                           const float* __restrict__ bias, float* __restrict__ C,
                           int M, int K, int Ncols) {
    long i = (long)blockIdx.x * blockDim.x + threadIdx.x;
    if (i >= (long)M * Ncols) return;
    int r = (int)(i / Ncols), c = (int)(i % Ncols);
    float acc = bias ? bias[c] : 0.f;
    const float* a = A + (size_t)r * K;
    for (int k = 0; k < K; k++) acc += a[k] * W[(size_t)k * Ncols + c];
    C[i] = acc;
}

__global__ void k_concat(const float* __restrict__ z, const float* __restrict__ desc,
                         float* __restrict__ zc) {
    long i = (long)blockIdx.x * blockDim.x + threadIdx.x;
    if (i >= (long)N_GRAPH * 328) return;
    int g = (int)(i / 328), c = (int)(i % 328);
    zc[i] = (c < 128) ? z[(size_t)g * 128 + c] : desc[(size_t)g * DESCN + (c - 128)];
}

__global__ void k_final(const float* __restrict__ z, const float* __restrict__ W,
                        const float* __restrict__ b, float* __restrict__ out) {
    int g = blockIdx.x * blockDim.x + threadIdx.x;
    if (g >= N_GRAPH) return;
    float acc = b[0];
    const float* zr = z + (size_t)g * 128;
#pragma unroll 4
    for (int k = 0; k < 128; k++) acc += zr[k] * W[k];
    out[g] = acc;
}

// ---------------- host orchestration ----------------
extern "C" void kernel_launch(void* const* d_in, const int* in_sizes, int n_in,
                              void* d_out, int out_size, void* d_ws, size_t ws_size,
                              hipStream_t stream) {
    (void)in_sizes; (void)n_in; (void)out_size; (void)ws_size;
    const float* x       = (const float*)d_in[0];
    const int*   ei      = (const int*)d_in[1];
    const int*   batch   = (const int*)d_in[2];
    const float* desc    = (const float*)d_in[3];
    const float* lin1_W  = (const float*)d_in[4];
    const float* lin1_b  = (const float*)d_in[5];
    const float* norm1_g = (const float*)d_in[6];
    const float* norm1_b = (const float*)d_in[7];
    const float* Wb      = (const float*)d_in[8];
    const float* Wc      = (const float*)d_in[9];
    const float* bcomb   = (const float*)d_in[10];
    const float* cbias   = (const float*)d_in[11];
    const float* norm_g  = (const float*)d_in[12];
    const float* norm_b  = (const float*)d_in[13];
    const float* mlp_W1  = (const float*)d_in[14];
    const float* mlp_g1  = (const float*)d_in[15];
    const float* mlp_b1  = (const float*)d_in[16];
    const float* mlp_W2  = (const float*)d_in[17];
    const float* mlp_g2  = (const float*)d_in[18];
    const float* mlp_b2  = (const float*)d_in[19];
    const float* lin2_W  = (const float*)d_in[20];
    const float* lin2_b  = (const float*)d_in[21];
    const float* bn2_g   = (const float*)d_in[22];
    const float* bn2_b   = (const float*)d_in[23];
    const float* out_W   = (const float*)d_in[24];
    const float* out_b   = (const float*)d_in[25];
    float* out = (float*)d_out;

    char* wsp = (char*)d_ws;
    size_t off = 0;
    auto alloc = [&](size_t bytes) -> void* {
        off = (off + 255) & ~(size_t)255;
        void* p = wsp + off;
        off += bytes;
        return p;
    };
    float* h     = (float*)alloc((size_t)N_NODES * HID * 4);
    bf16*  hb    = (bf16*) alloc((size_t)N_NODES * HID * 2);
    float* bs    = (float*)alloc((size_t)N_NODES * HID * 4);   // bases, then combine output m
    float* cb    = (float*)alloc((size_t)N_NODES * COMBN * 4);
    float* ssum  = (float*)alloc((size_t)N_NODES * HID * 4);
    float* smax  = (float*)alloc((size_t)N_NODES * HID * 4);
    float* cnt   = (float*)alloc((size_t)N_NODES * 4);
    bf16*  Wb_bf = (bf16*) alloc((size_t)NLAYER * HID * HID * 2);
    bf16*  Wc_bf = (bf16*) alloc((size_t)NLAYER * HID * COMBN * 2);
    float* part  = (float*)alloc((size_t)256 * 2 * HID * 4);
    float* stats = (float*)alloc((size_t)2 * HID * 4);
    float* gsum  = (float*)alloc((size_t)N_GRAPH * HID * 4);
    float* gcnt  = (float*)alloc((size_t)N_GRAPH * 4);
    float* z1    = (float*)alloc((size_t)N_GRAPH * 256 * 4);
    float* z2    = (float*)alloc((size_t)N_GRAPH * 128 * 4);
    float* zc    = (float*)alloc((size_t)N_GRAPH * 328 * 4);
    float* z3    = (float*)alloc((size_t)N_GRAPH * 128 * 4);

    auto blocks = [](long tot) { return (unsigned)((tot + 255) / 256); };

    // bf16 weight conversion (rerun each launch: no cross-call state allowed)
    { long n = (long)NLAYER * HID * HID;   k_cvt_bf16<<<blocks(n), 256, 0, stream>>>(Wb, Wb_bf, n); }
    { long n = (long)NLAYER * HID * COMBN; k_cvt_bf16<<<blocks(n), 256, 0, stream>>>(Wc, Wc_bf, n); }

    auto bn_block = [&](const float* xin, int rows, int cols, const float* gam, const float* bet,
                        const float* resid, float* yout, bf16* ybf) {
        int nblk = rows >= 32768 ? 256 : 64;
        k_bn_stats<<<nblk, 256, 0, stream>>>(xin, part, rows, cols);
        k_bn_fin<<<(cols + 255) / 256, 256, 0, stream>>>(part, stats, nblk, cols, 1.f / rows);
        long tot = (long)rows * cols;
        k_bn_apply<<<blocks(tot), 256, 0, stream>>>(xin, stats, gam, bet, resid, yout, ybf, tot, cols);
    };

    // input embedding: h = relu(bn(x @ lin1_W + b)); emit bf16 copy
    k_lin1<<<N_NODES, 256, 0, stream>>>(x, lin1_W, lin1_b, bs);
    bn_block(bs, N_NODES, HID, norm1_g, norm1_b, nullptr, h, hb);

    for (int l = 0; l < NLAYER; l++) {
        dim3 gb((N_NODES + 255) / 256, HID / 64);
        k_wmma_gemm_bf16<<<gb, 256, 0, stream>>>(hb, Wb_bf + (size_t)l * HID * HID,
                                                 nullptr, bs, N_NODES, HID, HID);
        dim3 gc((N_NODES + 255) / 256, COMBN / 64);
        k_wmma_gemm_bf16<<<gc, 256, 0, stream>>>(hb, Wc_bf + (size_t)l * HID * COMBN,
                                                 bcomb + l * COMBN, cb, N_NODES, COMBN, HID);
        { long tot = (long)N_NODES * HID;
          k_init_scatter<<<blocks(tot), 256, 0, stream>>>(ssum, smax, cnt); }
        { long tot = (long)(N_EDGES + N_NODES) * (HID / 4);
          k_scatter<<<blocks(tot), 256, 0, stream>>>(bs, ei, ssum, smax, cnt); }
        k_combine<<<(N_NODES + 3) / 4, 256, 0, stream>>>(ssum, smax, cnt, cb,
                                                         cbias + l * HID, bs);
        // h = h + relu(bn(m)); refresh bf16 copy except after last layer
        bn_block(bs, N_NODES, HID, norm_g + l * HID, norm_b + l * HID,
                 h, h, (l == NLAYER - 1) ? nullptr : hb);
    }

    // global mean pool
    { long tot = (long)N_GRAPH * HID; k_pool_init<<<blocks(tot), 256, 0, stream>>>(gsum, gcnt); }
    { long tot = (long)N_NODES * (HID / 4); k_pool<<<blocks(tot), 256, 0, stream>>>(h, batch, gsum, gcnt); }
    { long tot = (long)N_GRAPH * HID; k_pool_div<<<blocks(tot), 256, 0, stream>>>(gsum, gcnt); }

    // MLP tail on [4096, .]
    { long tot = (long)N_GRAPH * 256;
      k_gemm_f32<<<blocks(tot), 256, 0, stream>>>(gsum, mlp_W1, nullptr, z1, N_GRAPH, HID, 256); }
    bn_block(z1, N_GRAPH, 256, mlp_g1, mlp_b1, nullptr, z1, nullptr);
    { long tot = (long)N_GRAPH * 128;
      k_gemm_f32<<<blocks(tot), 256, 0, stream>>>(z1, mlp_W2, nullptr, z2, N_GRAPH, 256, 128); }
    bn_block(z2, N_GRAPH, 128, mlp_g2, mlp_b2, nullptr, z2, nullptr);
    { long tot = (long)N_GRAPH * 328;
      k_concat<<<blocks(tot), 256, 0, stream>>>(z2, desc, zc); }
    { long tot = (long)N_GRAPH * 128;
      k_gemm_f32<<<blocks(tot), 256, 0, stream>>>(zc, lin2_W, lin2_b, z3, N_GRAPH, 328, 128); }
    bn_block(z3, N_GRAPH, 128, bn2_g, bn2_b, nullptr, z3, nullptr);
    k_final<<<(N_GRAPH + 255) / 256, 256, 0, stream>>>(z3, out_W, out_b, out);
}
